// MambaBlock_28741921145336
// MI455X (gfx1250) — compile-verified, matrix operand order checked by reading, not run on hardware
//
#include <hip/hip_runtime.h>
#include <hip/hip_bf16.h>

// ---------------------------------------------------------------------------
// Problem constants (from reference)
// ---------------------------------------------------------------------------
#define Bsz      4
#define DIM      64
#define Lseq     16384            // H*W
#define BL       65536            // B*L
#define DINNER   128
#define DSTATE   16
#define DTRANK   4
#define XPROJ_N  36               // dt_rank + 2*d_state
#define XPROJ_NP 48               // padded to multiple of 16
#define HID      512
#define OUTD     128
#define ROWS2    32768            // B*DIM*H (rows for fc1/fc2)
#define NCHUNK   64               // scan chunks
#define CHLEN    256              // Lseq / NCHUNK

typedef __attribute__((ext_vector_type(16))) _Float16 v16h;
typedef __attribute__((ext_vector_type(8)))  _Float16 v8h;
typedef __attribute__((ext_vector_type(8)))  float    v8f;
typedef __attribute__((ext_vector_type(4)))  unsigned int u32x4;
typedef __attribute__((ext_vector_type(8)))  int      i32x8;
typedef __attribute__((ext_vector_type(4)))  int      i32x4;

#if defined(__gfx1250__) && __has_builtin(__builtin_amdgcn_tensor_load_to_lds)
#define USE_TDM 1
#else
#define USE_TDM 0
#endif

// ---------------------------------------------------------------------------
// K0: convert / pad weights to f16
// ---------------------------------------------------------------------------
__global__ void convert_weights(const float* __restrict__ ipw,   // (256,64)
                                const float* __restrict__ xpw,   // (36,128)
                                const float* __restrict__ opw,   // (64,128)
                                const float* __restrict__ f1w,   // (512,128)
                                const float* __restrict__ f2w,   // (128,512)
                                _Float16* __restrict__ ip16,
                                _Float16* __restrict__ xp16,     // padded (48,128)
                                _Float16* __restrict__ op16,
                                _Float16* __restrict__ f116,
                                _Float16* __restrict__ f216)
{
    int i = blockIdx.x * blockDim.x + threadIdx.x;
    if (i < 256 * 64) { ip16[i] = (_Float16)ipw[i]; return; }
    i -= 256 * 64;
    if (i < XPROJ_NP * 128) {
        int n = i >> 7, k = i & 127;
        xp16[i] = (_Float16)((n < XPROJ_N) ? xpw[n * 128 + k] : 0.0f);
        return;
    }
    i -= XPROJ_NP * 128;
    if (i < 64 * 128) { op16[i] = (_Float16)opw[i]; return; }
    i -= 64 * 128;
    if (i < HID * 128) { f116[i] = (_Float16)f1w[i]; return; }
    i -= HID * 128;
    if (i < OUTD * HID) { f216[i] = (_Float16)f2w[i]; return; }
}

// ---------------------------------------------------------------------------
// K1: LayerNorm over 64 channels; x is (B,64,H,W), output xn16 is (B*L, 64)
// ---------------------------------------------------------------------------
__global__ void ln_kernel(const float* __restrict__ x,
                          const float* __restrict__ gamma,
                          const float* __restrict__ beta,
                          _Float16* __restrict__ xn16)
{
    __shared__ float s1[64];
    __shared__ float s2[64];
    const int row = blockIdx.x;            // b*L + l
    const int c   = threadIdx.x;           // 0..63
    const int b   = row >> 14;
    const int l   = row & (Lseq - 1);
    float v = x[(((size_t)b * DIM + c) << 14) + l];
    s1[c] = v;
    s2[c] = v * v;
    __syncthreads();
    for (int st = 32; st > 0; st >>= 1) {
        if (c < st) { s1[c] += s1[c + st]; s2[c] += s2[c + st]; }
        __syncthreads();
    }
    const float mu  = s1[0] * (1.0f / 64.0f);
    const float var = s2[0] * (1.0f / 64.0f) - mu * mu;
    const float xnv = (v - mu) * rsqrtf(var + 1e-5f) * gamma[c] + beta[c];
    xn16[(size_t)row * 64 + c] = (_Float16)xnv;
}

// ---------------------------------------------------------------------------
// Generic WMMA GEMM:  C(M,N) = A(M,K) * W(N,K)^T
// Block = 8 waves computing a 128x16 macro-tile sharing one 16xK W tile in
// LDS, staged via the Tensor Data Mover (fallback: cooperative copy).
// Fragment layouts follow the documented CDNA5 16-bit striping:
//   lane l: r=l&15 (row), koff=(l>>4)*8; halves [kb+koff..+7], [kb+16+koff..+7]
//   C/D: lane l holds column n=l&15, VGPR r holds row m=(l>>4)*8+r
// MODE 0: plain f32 (ldc) | 1: out_proj transpose f16 | 2: bias+GELU f16
// MODE 3: bias f32
// ---------------------------------------------------------------------------
template <int MODE>
__global__ __launch_bounds__(256)
void gemm16(const _Float16* __restrict__ A,
            const _Float16* __restrict__ W,
            const float* __restrict__ bias,
            void* __restrict__ Cout,
            int M, int N, int K, int ldc)
{
    __shared__ _Float16 ldsW[16 * 512];    // max K = 512 -> 16KB
    const int lane   = threadIdx.x & 31;
    const int wib    = threadIdx.x >> 5;   // wave in block: 0..7
    const int tilesN = N >> 4;
    const int tn     = (blockIdx.x % tilesN) << 4;
    const int tm     = (blockIdx.x / tilesN) * 128 + (wib << 4);
    if (tm >= M) return;
    const int half = lane >> 4;
    const int lr   = lane & 15;
    const int koff = half << 3;

    // ---- stage W tile rows [tn, tn+16) x K into LDS -----------------------
#if USE_TDM
    if (wib == 0) {
        const unsigned long long gaddr =
            (unsigned long long)(const void*)(W + (size_t)tn * K);
        const unsigned int laddr = (unsigned int)(uintptr_t)(void*)&ldsW[0];
        const unsigned int total = (unsigned int)(16 * K);  // f16 elements
        // D# group 0: count=1 | lds_addr | global_addr[56:0] | type=2
        u32x4 g0 = { 1u, laddr,
                     (unsigned int)(gaddr & 0xFFFFFFFFu),
                     (unsigned int)((gaddr >> 32) & 0x01FFFFFFu) | (2u << 30) };
        // D# group 1: data_size=1(2B); tensor_dim0=total; tensor_dim1=1;
        //             tile_dim0=total; tile_dim1=1; dim0_stride=total
        i32x8 g1 = { (int)(1u << 16),
                     (int)((total & 0xFFFFu) << 16),
                     (int)(((total >> 16) & 0xFFFFu) | (1u << 16)),
                     (int)((total & 0xFFFFu) << 16),
                     (int)1,
                     (int)total,
                     0, 0 };
        i32x4 gz4 = { 0, 0, 0, 0 };
        i32x8 gz8 = { 0, 0, 0, 0, 0, 0, 0, 0 };
        __builtin_amdgcn_tensor_load_to_lds(g0, g1, gz4, gz4, gz8, 0);
        __builtin_amdgcn_s_wait_tensorcnt(0);
    }
    __syncthreads();
#else
    for (int i = threadIdx.x; i < (16 * K) / 8; i += 256)
        ((v8h*)ldsW)[i] = ((const v8h*)(W + (size_t)tn * K))[i];
    __syncthreads();
#endif

    const _Float16* aRow = A + (size_t)(tm + lr) * K;
    const _Float16* wRow = ldsW + (size_t)lr * K;

    v8f acc = {};
    for (int kb = 0; kb < K; kb += 32) {
        v16h a, b;
        *(v8h*)&a       = *(const v8h*)(aRow + kb + koff);
        *((v8h*)&a + 1) = *(const v8h*)(aRow + kb + koff + 16);
        *(v8h*)&b       = *(const v8h*)(wRow + kb + koff);
        *((v8h*)&b + 1) = *(const v8h*)(wRow + kb + koff + 16);
        acc = __builtin_amdgcn_wmma_f32_16x16x32_f16(
            false, a, false, b, (short)0, acc, false, false);
    }

    const int n     = lr;
    const int mbase = half << 3;
#pragma unroll
    for (int r = 0; r < 8; ++r) {
        const int m = tm + mbase + r;
        float v = acc[r];
        if (MODE == 0) {
            ((float*)Cout)[(size_t)m * ldc + tn + n] = v;
        } else if (MODE == 1) {
            const int bi = m >> 14;
            const int l  = m & (Lseq - 1);
            const int c  = tn + n;
            ((_Float16*)Cout)[(((size_t)(bi * DIM + c)) << 14) + l] = (_Float16)v;
        } else if (MODE == 2) {
            v += bias[tn + n];
            const float g = 0.5f * v * (1.0f + erff(v * 0.70710678118f));
            ((_Float16*)Cout)[(size_t)m * ldc + tn + n] = (_Float16)g;
        } else {
            v += bias[tn + n];
            ((float*)Cout)[(size_t)m * ldc + tn + n] = v;
        }
    }
}

// ---------------------------------------------------------------------------
// K3: split xz -> (xi, z); depthwise causal conv(4) + SiLU on xi
// ---------------------------------------------------------------------------
__global__ void conv_silu_kernel(const float* __restrict__ xz,     // (BL,256)
                                 const float* __restrict__ cw,     // (128,4)
                                 const float* __restrict__ cb,     // (128)
                                 float* __restrict__ xs32,         // (BL,128)
                                 _Float16* __restrict__ xs16,      // (BL,128)
                                 _Float16* __restrict__ z16)       // (BL,128)
{
    const int idx = blockIdx.x * blockDim.x + threadIdx.x;
    if (idx >= BL * DINNER) return;
    const int d   = idx & (DINNER - 1);
    const int row = idx >> 7;               // b*L + l
    const int l   = row & (Lseq - 1);
    float acc = cb[d];
#pragma unroll
    for (int j = 0; j < 4; ++j) {
        const int ll = l - 3 + j;
        if (ll >= 0) acc += cw[d * 4 + j] * xz[(size_t)(row - 3 + j) * 256 + d];
    }
    const float s = acc / (1.0f + __expf(-acc));   // SiLU
    xs32[idx] = s;
    xs16[idx] = (_Float16)s;
    z16[idx]  = (_Float16)xz[(size_t)row * 256 + 128 + d];
}

// ---------------------------------------------------------------------------
// K5: dt = softplus(dt_low @ dt_proj_w^T + dt_proj_b)
// ---------------------------------------------------------------------------
__global__ void dt_kernel(const float* __restrict__ xdbl,   // (BL,48)
                          const float* __restrict__ dpw,    // (128,4)
                          const float* __restrict__ dpb,    // (128)
                          float* __restrict__ dtf)          // (BL,128)
{
    const int idx = blockIdx.x * blockDim.x + threadIdx.x;
    if (idx >= BL * DINNER) return;
    const int d   = idx & (DINNER - 1);
    const int row = idx >> 7;
    float acc = dpb[d];
#pragma unroll
    for (int r = 0; r < DTRANK; ++r)
        acc += xdbl[(size_t)row * XPROJ_NP + r] * dpw[d * DTRANK + r];
    dtf[idx] = (acc > 20.0f) ? acc : log1pf(__expf(acc));
}

// ---------------------------------------------------------------------------
// Chunked selective scan. Lane = (b,d,s); wave32 = 2 channels x 16 states.
// Wave id decomposition: w = ((b*64 + dpair)*NCHUNK + c)
// Pass 1: per chunk, h0=0: P = prod(dA), q = local final h  (16384 waves)
// Combine: per (b,d,s) lane, 64-step sequential fix -> chunk-initial h0
// Pass 2: replay chunk with true h0, emit y via 16-lane shuffle reduce
// ---------------------------------------------------------------------------
__device__ __forceinline__ void scan_decomp(int tid, int lane,
                                            int& b, int& d, int& s, int& c)
{
    s = lane & 15;
    const int dlow = lane >> 4;
    const int w    = tid >> 5;
    c = w & (NCHUNK - 1);
    const int rest = w >> 6;
    b = rest >> 6;
    d = ((rest & 63) << 1) | dlow;
}

__global__ __launch_bounds__(256)
void scan_pass1(const float* __restrict__ dtf,    // (BL,128)
                const float* __restrict__ xdbl,   // (BL,48)
                const float* __restrict__ xs32,   // (BL,128)
                const float* __restrict__ A_log,  // (128,16)
                float* __restrict__ carryP,       // (8192, NCHUNK)
                float* __restrict__ carryQ)       // (8192, NCHUNK)
{
    const int tid  = blockIdx.x * blockDim.x + threadIdx.x;
    const int lane = threadIdx.x & 31;
    int b, d, s, c;
    scan_decomp(tid, lane, b, d, s, c);
    const float Aval = -__expf(A_log[d * DSTATE + s]);
    float P = 1.0f, h = 0.0f;
    const size_t base = ((size_t)b << 14) + (size_t)c * CHLEN;
    for (int t = 0; t < CHLEN; ++t) {
        const size_t row = base + t;
        if (t + 8 < CHLEN)
            __builtin_prefetch(&dtf[(row + 8) * DINNER + d], 0, 1);
        const float dt = dtf[row * DINNER + d];
        const float Bv = xdbl[row * XPROJ_NP + DTRANK + s];
        const float xv = xs32[row * DINNER + d];
        const float dA = __expf(dt * Aval);
        P *= dA;
        h = dA * h + (dt * xv) * Bv;
    }
    const size_t idx = ((size_t)((b * DINNER + d) * DSTATE + s)) * NCHUNK + c;
    carryP[idx] = P;
    carryQ[idx] = h;
}

__global__ void scan_combine(const float* __restrict__ carryP,
                             const float* __restrict__ carryQ,
                             float* __restrict__ carryH)
{
    const int i = blockIdx.x * blockDim.x + threadIdx.x;   // 0..8191 (b,d,s)
    if (i >= Bsz * DINNER * DSTATE) return;
    float h = 0.0f;
    for (int c = 0; c < NCHUNK; ++c) {
        const size_t idx = (size_t)i * NCHUNK + c;
        carryH[idx] = h;
        h = carryP[idx] * h + carryQ[idx];
    }
}

__global__ __launch_bounds__(256)
void scan_pass2(const float* __restrict__ dtf,
                const float* __restrict__ xdbl,
                const float* __restrict__ xs32,
                const float* __restrict__ A_log,
                const float* __restrict__ carryH,
                float* __restrict__ y32)          // (BL,128)
{
    const int tid  = blockIdx.x * blockDim.x + threadIdx.x;
    const int lane = threadIdx.x & 31;
    int b, d, s, c;
    scan_decomp(tid, lane, b, d, s, c);
    const float Aval = -__expf(A_log[d * DSTATE + s]);
    float h = carryH[((size_t)((b * DINNER + d) * DSTATE + s)) * NCHUNK + c];
    const size_t base = ((size_t)b << 14) + (size_t)c * CHLEN;
    for (int t = 0; t < CHLEN; ++t) {
        const size_t row = base + t;
        if (t + 8 < CHLEN)
            __builtin_prefetch(&xdbl[(row + 8) * XPROJ_NP + DTRANK + s], 0, 1);
        const float dt = dtf[row * DINNER + d];
        const float Bv = xdbl[row * XPROJ_NP + DTRANK + s];
        const float Cv = xdbl[row * XPROJ_NP + DTRANK + DSTATE + s];
        const float xv = xs32[row * DINNER + d];
        const float dA = __expf(dt * Aval);
        h = dA * h + (dt * xv) * Bv;
        float p = h * Cv;
        p += __shfl_xor(p, 1, 16);
        p += __shfl_xor(p, 2, 16);
        p += __shfl_xor(p, 4, 16);
        p += __shfl_xor(p, 8, 16);
        if (s == 0) y32[row * DINNER + d] = p;
    }
}

// ---------------------------------------------------------------------------
// K7: gating:  ym = (y + xs*D) * silu(z)   -> f16 for out_proj GEMM
// ---------------------------------------------------------------------------
__global__ void gate_kernel(const float* __restrict__ y32,
                            const float* __restrict__ xs32,
                            const float* __restrict__ Dv,
                            const _Float16* __restrict__ z16,
                            _Float16* __restrict__ ym16)
{
    const int idx = blockIdx.x * blockDim.x + threadIdx.x;
    if (idx >= BL * DINNER) return;
    const int d = idx & (DINNER - 1);
    const float yv = y32[idx] + xs32[idx] * Dv[d];
    const float zv = (float)z16[idx];
    const float sz = zv / (1.0f + __expf(-zv));
    ym16[idx] = (_Float16)(yv * sz);
}

// ---------------------------------------------------------------------------
// Host launcher
// ---------------------------------------------------------------------------
extern "C" void kernel_launch(void* const* d_in, const int* in_sizes, int n_in,
                              void* d_out, int out_size, void* d_ws, size_t ws_size,
                              hipStream_t stream)
{
    const float* x         = (const float*)d_in[0];
    const float* ln_g      = (const float*)d_in[1];
    const float* ln_b      = (const float*)d_in[2];
    const float* in_proj_w = (const float*)d_in[3];
    const float* conv_w    = (const float*)d_in[4];
    const float* conv_b    = (const float*)d_in[5];
    const float* x_proj_w  = (const float*)d_in[6];
    const float* dt_proj_w = (const float*)d_in[7];
    const float* dt_proj_b = (const float*)d_in[8];
    const float* A_log     = (const float*)d_in[9];
    const float* Dvec      = (const float*)d_in[10];
    const float* out_prj_w = (const float*)d_in[11];
    const float* fc1_w     = (const float*)d_in[12];
    const float* fc1_b     = (const float*)d_in[13];
    const float* fc2_w     = (const float*)d_in[14];
    const float* fc2_b     = (const float*)d_in[15];

    // -------- workspace layout (recomputed identically every call) --------
    char* wsp = (char*)d_ws;
    auto alloc = [&](size_t bytes) -> char* {
        char* p = wsp;
        wsp += (bytes + 255) & ~(size_t)255;
        return p;
    };
    _Float16* ip16  = (_Float16*)alloc((size_t)256 * 64 * 2);
    _Float16* xp16  = (_Float16*)alloc((size_t)XPROJ_NP * 128 * 2);
    _Float16* op16  = (_Float16*)alloc((size_t)64 * 128 * 2);
    _Float16* f116  = (_Float16*)alloc((size_t)HID * 128 * 2);
    _Float16* f216  = (_Float16*)alloc((size_t)OUTD * HID * 2);
    _Float16* xn16  = (_Float16*)alloc((size_t)BL * 64 * 2);
    float*    xz32  = (float*)   alloc((size_t)BL * 256 * 4);   // aliased as y32 later
    float*    xs32  = (float*)   alloc((size_t)BL * 128 * 4);
    _Float16* xs16  = (_Float16*)alloc((size_t)BL * 128 * 2);   // aliased as ym16 later
    _Float16* z16   = (_Float16*)alloc((size_t)BL * 128 * 2);
    float*    xdbl  = (float*)   alloc((size_t)BL * XPROJ_NP * 4);
    float*    dtf   = (float*)   alloc((size_t)BL * 128 * 4);
    _Float16* out16 = (_Float16*)alloc((size_t)ROWS2 * 128 * 2);
    _Float16* h116  = (_Float16*)alloc((size_t)ROWS2 * HID * 2);
    float*    carP  = (float*)   alloc((size_t)8192 * NCHUNK * 4);
    float*    carQ  = (float*)   alloc((size_t)8192 * NCHUNK * 4);
    float*    carH  = (float*)   alloc((size_t)8192 * NCHUNK * 4);
    float*    y32   = xz32;        // xz dead after conv split
    _Float16* ym16  = xs16;        // xs16 dead after x_proj GEMM

    // -------- pipeline --------
    {   // K0: weight conversion
        const int n = 256*64 + XPROJ_NP*128 + 64*128 + HID*128 + OUTD*HID;
        convert_weights<<<(n + 255) / 256, 256, 0, stream>>>(
            in_proj_w, x_proj_w, out_prj_w, fc1_w, fc2_w,
            ip16, xp16, op16, f116, f216);
    }
    // K1: layernorm
    ln_kernel<<<BL, 64, 0, stream>>>(x, ln_g, ln_b, xn16);
    // K2: in_proj  (65536 x 64) @ (64 x 256) -> xz32   [grid = (M/128)*tilesN]
    gemm16<0><<<512 * 16, 256, 0, stream>>>(
        xn16, ip16, nullptr, xz32, BL, 256, 64, 256);
    // K3: split + depthwise conv + SiLU
    conv_silu_kernel<<<(BL * DINNER) / 256, 256, 0, stream>>>(
        xz32, conv_w, conv_b, xs32, xs16, z16);
    // K4: x_proj  (65536 x 128) @ (128 x 48pad) -> xdbl
    gemm16<0><<<512 * 3, 256, 0, stream>>>(
        xs16, xp16, nullptr, xdbl, BL, XPROJ_NP, 128, XPROJ_NP);
    // K5: dt projection + softplus
    dt_kernel<<<(BL * DINNER) / 256, 256, 0, stream>>>(
        xdbl, dt_proj_w, dt_proj_b, dtf);
    // K6: chunked selective scan (pass1 / combine / pass2)
    scan_pass1<<<(Bsz * DINNER * DSTATE * NCHUNK) / 256, 256, 0, stream>>>(
        dtf, xdbl, xs32, A_log, carP, carQ);
    scan_combine<<<(Bsz * DINNER * DSTATE) / 256, 256, 0, stream>>>(
        carP, carQ, carH);
    scan_pass2<<<(Bsz * DINNER * DSTATE * NCHUNK) / 256, 256, 0, stream>>>(
        dtf, xdbl, xs32, A_log, carH, y32);
    // K7: gate
    gate_kernel<<<(BL * DINNER) / 256, 256, 0, stream>>>(
        y32, xs32, Dvec, z16, ym16);
    // K8: out_proj (65536 x 128) @ (128 x 64) -> out16 in (B,64,H,W) layout
    gemm16<1><<<512 * 4, 256, 0, stream>>>(
        ym16, op16, nullptr, out16, BL, 64, 128, 0);
    // K9: fc1 + GELU  (32768 x 128) @ (128 x 512) -> h1 f16
    gemm16<2><<<256 * 32, 256, 0, stream>>>(
        out16, f116, fc1_b, h116, ROWS2, HID, 128, HID);
    // K10: fc2 + bias (32768 x 512) @ (512 x 128) -> d_out f32
    gemm16<3><<<256 * 8, 256, 0, stream>>>(
        h116, f216, fc2_b, (float*)d_out, ROWS2, OUTD, HID, OUTD);
}
